// RPN_31516470018587
// MI455X (gfx1250) — matrix operand order repaired and checked by reading, not compiled, not run on hardware
//
#include <hip/hip_runtime.h>
#include <hip/hip_bf16.h>
#include <math.h>

typedef __bf16 bf16_t;
typedef bf16_t v16bf __attribute__((ext_vector_type(16)));
typedef bf16_t v8bf  __attribute__((ext_vector_type(8)));
typedef float  v8f   __attribute__((ext_vector_type(8)));

#define PRE_NMS 400
#define POST_NMS 100
#define NMS_TH 0.7f
#define SCALE_CLAMP_F 3.3322045f  /* ln(224/8) */

__device__ __forceinline__ unsigned f32_to_bf16_rne(float f) {
  union { float f; unsigned u; } v; v.f = f;
  unsigned r = v.u + 0x7FFFu + ((v.u >> 16) & 1u);
  return r >> 16;
}

__device__ __forceinline__ v16bf frag_from_lds(const unsigned short* p,
                                               int o0, int o1) {
  v8bf lo = *(const v8bf*)(p + o0);
  v8bf hi = *(const v8bf*)(p + o1);
  v16bf r;
#pragma unroll
  for (int i = 0; i < 8; ++i) { r[i] = lo[i]; r[i + 8] = hi[i]; }
  return r;
}

// ---------------------------------------------------------------------------
// w_stem OIHW f32 [co][ci][ky][kx] -> bf16 [tap][co][ci]
__global__ void reorg_weights(const float* __restrict__ w,
                              unsigned short* __restrict__ wB, int total) {
  int i = blockIdx.x * blockDim.x + threadIdx.x;
  if (i >= total) return;
  int tap = i / 65536;
  int rem = i - tap * 65536;
  int co = rem >> 8;
  int ci = rem & 255;
  wB[i] = (unsigned short)f32_to_bf16_rne(w[((co << 8) + ci) * 9 + tap]);
}

// scores[i] = b_obj[a]; deltas[i][0..3] = b_box[a*4+d]   (i over B*HW*3)
__global__ void init_head(float* __restrict__ scores, float* __restrict__ deltas,
                          const float* __restrict__ b_obj,
                          const float* __restrict__ b_box, int total3) {
  int i = blockIdx.x * blockDim.x + threadIdx.x;
  if (i >= total3) return;
  int a = i % 3;
  scores[i] = b_obj[a];
  float4 d;
  d.x = b_box[a * 4 + 0]; d.y = b_box[a * 4 + 1];
  d.z = b_box[a * 4 + 2]; d.w = b_box[a * 4 + 3];
  ((float4*)deltas)[i] = d;
}

// ---------------------------------------------------------------------------
// Fused 3x3 stem conv (implicit GEMM on WMMA bf16) + bias + ReLU + 1x1 heads.
// Block tile: M=64 spatial x N=128 out-channels, 256 threads (8 wave32).
// Wave tile: 32x32 -> 4 WMMAs per K-step from 2 A + 2 B fragments.
__global__ __launch_bounds__(256) void stem_gemm_fused(
    const float* __restrict__ feat,           // [B,256,H,W] f32
    const unsigned short* __restrict__ wB,    // [9][256 co][256 ci] bf16
    const float* __restrict__ b_stem,         // [256]
    const float* __restrict__ w_obj,          // [3][256]
    const float* __restrict__ w_box,          // [12][256]
    float* __restrict__ scores,               // [B*HW*3]   (bias pre-init)
    float* __restrict__ deltas,               // [B*HW*3*4] (bias pre-init)
    int H, int W) {
  const int HW   = H * W;
  const int tid  = threadIdx.x;
  const int wave = tid >> 5;
  const int lane = tid & 31;
  const int m0   = blockIdx.x * 64;           // global row (b*HW + sp)
  const int b    = m0 / HW;                   // HW is a multiple of 64
  const int sp0  = m0 - b * HW;
  const int nBase = blockIdx.y * 128;
  const int mw = (wave & 1) * 32;             // wave M offset in tile
  const int nw = (wave >> 1) * 32;            // wave N offset in tile

  __shared__ __align__(16) unsigned short ldsA[64 * 40];   // [m][k] bf16
  __shared__ __align__(16) unsigned short ldsB[128 * 40];  // [n][k] bf16
  __shared__ __align__(16) float ldsS[64 * 132];           // [m][n] stem f32

  v8f acc00 = {}, acc01 = {}, acc10 = {}, acc11 = {};

  const int row = lane & 15;
  const int ahk = (lane >> 4) << 3;   // A k-offset: 0 or 8
  const int bk  = (lane >> 4) << 3;   // B k-offset: 0 or 8

  // A-loader mapping: this thread owns spatial row ml (0..63) and 8 channels
  const int ml = tid & 63;
  const int cg = (tid >> 6) << 3;               // 0,8,16,24
  const int sp = sp0 + ml;
  const int y0 = sp / W;
  const int x0 = sp - y0 * W;
  const float* featB = feat + (size_t)b * 256 * (size_t)HW;
  unsigned* ldsAdst = (unsigned*)&ldsA[ml * 40 + cg];     // 16B aligned

  // B-loader mapping: two 16B async copies per thread (128 rows x 32 ci)
  const int nl = tid >> 1;                      // 0..127
  const int part = (tid & 1) << 4;              // 0 or 16 (bf16 elements)
  const unsigned ldsBoff0 =
      (unsigned)(uintptr_t)(unsigned short*)&ldsB[nl * 40 + part];
  const unsigned ldsBoff1 = ldsBoff0 + 16;

  for (int tap = 0; tap < 9; ++tap) {
    const int ky = tap / 3 - 1;
    const int kx = tap - (tap / 3) * 3 - 1;
    const int y = y0 + ky;
    const int x = x0 + kx;
    const bool valid = ((unsigned)x < (unsigned)W) && ((unsigned)y < (unsigned)H);
    const int offs = valid ? (y * W + x) : 0;   // safe in-bounds offset

    for (int c0 = 0; c0 < 256; c0 += 32) {
      __syncthreads();
      // ---- B tile: 128 co x 32 ci via async copies straight into LDS
      {
        const unsigned short* src =
            wB + (((size_t)tap * 256 + nBase + nl) << 8) + c0 + part;
        asm volatile("global_load_async_to_lds_b128 %0, %1, off"
                     :: "v"(ldsBoff0),
                        "v"((unsigned long long)(uintptr_t)src)
                     : "memory");
        asm volatile("global_load_async_to_lds_b128 %0, %1, off"
                     :: "v"(ldsBoff1),
                        "v"((unsigned long long)(uintptr_t)(src + 8))
                     : "memory");
      }
      // ---- A tile: one m, 8 channels; branch-free bounds via safe offset
      {
        const float* fp = featB + ((size_t)(c0 + cg)) * (size_t)HW + offs;
        float f[8];
#pragma unroll
        for (int i = 0; i < 8; ++i) f[i] = fp[(size_t)i * HW];
        const float zmask = valid ? 1.0f : 0.0f;
        unsigned p[4];
#pragma unroll
        for (int i = 0; i < 4; ++i) {
          unsigned lo = f32_to_bf16_rne(f[2 * i] * zmask);
          unsigned hi = f32_to_bf16_rne(f[2 * i + 1] * zmask);
          p[i] = lo | (hi << 16);
        }
        ldsAdst[0] = p[0]; ldsAdst[1] = p[1];
        ldsAdst[2] = p[2]; ldsAdst[3] = p[3];
      }
      asm volatile("s_wait_asynccnt 0" ::: "memory");
      __syncthreads();

      // ---- fragments per ISA wave32 layouts
      v16bf a0 = frag_from_lds(&ldsA[(mw + row) * 40], ahk, ahk + 16);
      v16bf a1 = frag_from_lds(&ldsA[(mw + 16 + row) * 40], ahk, ahk + 16);
      v16bf b0 = frag_from_lds(&ldsB[(nw + row) * 40], bk, bk + 16);
      v16bf b1 = frag_from_lds(&ldsB[(nw + 16 + row) * 40], bk, bk + 16);

      acc00 = __builtin_amdgcn_wmma_f32_16x16x32_bf16(
          false, a0, false, b0, (short)0, acc00, false, false);
      acc01 = __builtin_amdgcn_wmma_f32_16x16x32_bf16(
          false, a0, false, b1, (short)0, acc01, false, false);
      acc10 = __builtin_amdgcn_wmma_f32_16x16x32_bf16(
          false, a1, false, b0, (short)0, acc10, false, false);
      acc11 = __builtin_amdgcn_wmma_f32_16x16x32_bf16(
          false, a1, false, b1, (short)0, acc11, false, false);
    }
  }

  // ---- epilogue: bias + ReLU -> LDS stem tile (C/D: lanes 16-31 -> M+8)
  __syncthreads();
  {
    const int mo = (lane >> 4) << 3;
    const int nn = lane & 15;
    const float bia0 = b_stem[nBase + nw + nn];
    const float bia1 = b_stem[nBase + nw + nn + 16];
#pragma unroll
    for (int r = 0; r < 8; ++r) {
      const int ma = mw + mo + r;
      const int mb = mw + 16 + mo + r;
      float v;
      v = acc00[r] + bia0; ldsS[ma * 132 + nw + nn]      = v > 0.f ? v : 0.f;
      v = acc01[r] + bia1; ldsS[ma * 132 + nw + nn + 16] = v > 0.f ? v : 0.f;
      v = acc10[r] + bia0; ldsS[mb * 132 + nw + nn]      = v > 0.f ? v : 0.f;
      v = acc11[r] + bia1; ldsS[mb * 132 + nw + nn + 16] = v > 0.f ? v : 0.f;
    }
  }
  __syncthreads();

  // ---- fused 1x1 heads: partial dots over this block's 128 channels
  for (int idx = tid; idx < 64 * 15; idx += 256) {
    const int m = idx / 15;
    const int o = idx - m * 15;
    const float* wrow =
        (o < 3 ? (w_obj + o * 256) : (w_box + (o - 3) * 256)) + nBase;
    float s = 0.f;
#pragma unroll 8
    for (int n = 0; n < 128; ++n) s += ldsS[m * 132 + n] * wrow[n];
    const int mg = m0 + m;
    if (o < 3) {
      atomicAdd(&scores[mg * 3 + o], s);
    } else {
      const int q = o - 3;
      atomicAdd(&deltas[(mg * 3 + (q >> 2)) * 4 + (q & 3)], s);
    }
  }
}

// ---------------------------------------------------------------------------
// Anchor decode + delta application + clip
__global__ void decode_boxes(const float* __restrict__ deltas,
                             float* __restrict__ boxes,
                             const int* __restrict__ strideP,
                             const int* __restrict__ stride3P,
                             int HW, int W, int total3, int H3) {
  int i = blockIdx.x * blockDim.x + threadIdx.x;
  if (i >= total3) return;
  const float stride = (float)(*strideP);
  const float img = (float)H3 * (float)(*stride3P);
  const int a = i % 3;
  const int m = (i / 3) % HW;
  const int x = m % W, y = m / W;
  const float xc = (x + 0.5f) * stride, yc = (y + 0.5f) * stride;
  const float area = 64.0f * stride * stride;       // (8*stride)^2
  const float ar = (a == 0) ? 0.5f : (a == 1) ? 1.0f : 2.0f;
  const float wa = sqrtf(area / ar);
  const float ha = area / wa;
  const float dx = deltas[i * 4 + 0], dy = deltas[i * 4 + 1];
  const float dw = fminf(deltas[i * 4 + 2], SCALE_CLAMP_F);
  const float dh = fminf(deltas[i * 4 + 3], SCALE_CLAMP_F);
  const float pcx = dx * wa + xc, pcy = dy * ha + yc;
  const float pw = expf(dw) * wa, ph = expf(dh) * ha;
  boxes[i * 4 + 0] = fminf(fmaxf(pcx - 0.5f * pw, 0.f), img);
  boxes[i * 4 + 1] = fminf(fmaxf(pcy - 0.5f * ph, 0.f), img);
  boxes[i * 4 + 2] = fminf(fmaxf(pcx + 0.5f * pw, 0.f), img);
  boxes[i * 4 + 3] = fminf(fmaxf(pcy + 0.5f * ph, 0.f), img);
}

// ---------------------------------------------------------------------------
// Top-400 per batch via iterative argmax with owner-rescan (sorted output).
__global__ __launch_bounds__(1024) void topk400(
    float* __restrict__ scores, const float* __restrict__ boxes, int N,
    float* __restrict__ tkS, float* __restrict__ tkB) {
  const int b = blockIdx.x;
  float* sc = scores + (size_t)b * N;
  const float* bx = boxes + (size_t)b * N * 4;
  const int tid = threadIdx.x;
  __shared__ float sval[1024];
  __shared__ int   sidx[1024];
  __shared__ float rval[1024];
  __shared__ int   ridx[1024];

  float mv = -INFINITY; int mi = 0;
  for (int i = tid; i < N; i += 1024) {
    float v = sc[i];
    if (v > mv) { mv = v; mi = i; }
  }
  sval[tid] = mv; sidx[tid] = mi;

  for (int it = 0; it < PRE_NMS; ++it) {
    __syncthreads();
    rval[tid] = sval[tid]; ridx[tid] = sidx[tid];
    __syncthreads();
    for (int offw = 512; offw > 0; offw >>= 1) {
      if (tid < offw && rval[tid + offw] > rval[tid]) {
        rval[tid] = rval[tid + offw];
        ridx[tid] = ridx[tid + offw];
      }
      __syncthreads();
    }
    const int win = ridx[0];
    if (tid == 0) {
      tkS[b * PRE_NMS + it] = rval[0];
      tkB[(b * PRE_NMS + it) * 4 + 0] = bx[win * 4 + 0];
      tkB[(b * PRE_NMS + it) * 4 + 1] = bx[win * 4 + 1];
      tkB[(b * PRE_NMS + it) * 4 + 2] = bx[win * 4 + 2];
      tkB[(b * PRE_NMS + it) * 4 + 3] = bx[win * 4 + 3];
    }
    if ((win & 1023) == tid) {          // owner invalidates and rescans chunk
      sc[win] = -INFINITY;
      float m2 = -INFINITY; int i2 = 0;
      for (int i = tid; i < N; i += 1024) {
        float v = sc[i];
        if (v > m2) { m2 = v; i2 = i; }
      }
      sval[tid] = m2; sidx[tid] = i2;
    }
  }
}

// ---------------------------------------------------------------------------
// Greedy NMS over 400 sorted boxes; first <=100 kept -> out, zero-padded.
__global__ __launch_bounds__(512) void nms100(const float* __restrict__ tkB,
                                              float* __restrict__ out,
                                              int level) {
  const int b = blockIdx.x;
  const int tid = threadIdx.x;
  __shared__ float bx0[PRE_NMS], by0[PRE_NMS], bx1[PRE_NMS], by1[PRE_NMS];
  __shared__ float barea[PRE_NMS];
  __shared__ int keep[PRE_NMS];
  for (int i = tid; i < PRE_NMS; i += 512) {
    float x1 = tkB[(b * PRE_NMS + i) * 4 + 0];
    float y1 = tkB[(b * PRE_NMS + i) * 4 + 1];
    float x2 = tkB[(b * PRE_NMS + i) * 4 + 2];
    float y2 = tkB[(b * PRE_NMS + i) * 4 + 3];
    bx0[i] = x1; by0[i] = y1; bx1[i] = x2; by1[i] = y2;
    barea[i] = (x2 - x1) * (y2 - y1);
    keep[i] = 1;
  }
  __syncthreads();
  for (int i = 0; i < PRE_NMS; ++i) {
    if (keep[i]) {
      const float ix0 = bx0[i], iy0 = by0[i], ix1 = bx1[i], iy1 = by1[i];
      const float ia = barea[i];
      for (int j = i + 1 + tid; j < PRE_NMS; j += 512) {
        float lx = fmaxf(ix0, bx0[j]);
        float ly = fmaxf(iy0, by0[j]);
        float rx = fminf(ix1, bx1[j]);
        float ry = fminf(iy1, by1[j]);
        float iw = fmaxf(rx - lx, 0.f), ih = fmaxf(ry - ly, 0.f);
        float inter = iw * ih;
        float uni = fmaxf(ia + barea[j] - inter, 1e-8f);
        if (inter / uni > NMS_TH) keep[j] = 0;
      }
    }
    __syncthreads();
  }
  if (tid == 0) {
    int cnt = 0;
    float* dst = out + ((size_t)b * 300 + (size_t)level * 100) * 4;
    for (int i = 0; i < PRE_NMS && cnt < POST_NMS; ++i) {
      if (keep[i]) {
        dst[cnt * 4 + 0] = bx0[i];
        dst[cnt * 4 + 1] = by0[i];
        dst[cnt * 4 + 2] = bx1[i];
        dst[cnt * 4 + 3] = by1[i];
        ++cnt;
      }
    }
    for (; cnt < POST_NMS; ++cnt) {
      dst[cnt * 4 + 0] = 0.f; dst[cnt * 4 + 1] = 0.f;
      dst[cnt * 4 + 2] = 0.f; dst[cnt * 4 + 3] = 0.f;
    }
  }
}

// ---------------------------------------------------------------------------
extern "C" void kernel_launch(void* const* d_in, const int* in_sizes, int n_in,
                              void* d_out, int out_size, void* d_ws,
                              size_t ws_size, hipStream_t stream) {
  (void)n_in; (void)out_size; (void)ws_size;
  const float* feats[3] = {(const float*)d_in[0], (const float*)d_in[1],
                           (const float*)d_in[2]};
  const float* w_stem = (const float*)d_in[3];
  const float* b_stem = (const float*)d_in[4];
  const float* w_obj  = (const float*)d_in[5];
  const float* b_obj  = (const float*)d_in[6];
  const float* w_box  = (const float*)d_in[7];
  const float* b_box  = (const float*)d_in[8];
  const int* strides[3] = {(const int*)d_in[9], (const int*)d_in[10],
                           (const int*)d_in[11]};

  const int B = 4, C = 256;
  int Hs[3], Ws[3];
  for (int l = 0; l < 3; ++l) {
    long hw = (long)in_sizes[l] / (B * C);
    int w = (int)(sqrt((double)hw) + 0.5);
    Ws[l] = w;
    Hs[l] = (int)(hw / w);
  }

  size_t off = 0;
  auto carve = [&](size_t bytes) -> void* {
    void* p = (char*)d_ws + off;
    off += (bytes + 255) & ~(size_t)255;
    return p;
  };
  unsigned short* wB = (unsigned short*)carve(9ull * 256 * 256 * 2);
  const size_t HWmax = (size_t)Hs[0] * Ws[0];
  float* scores = (float*)carve((size_t)B * HWmax * 3 * 4);
  float* deltas = (float*)carve((size_t)B * HWmax * 3 * 4 * 4);
  float* boxes  = (float*)carve((size_t)B * HWmax * 3 * 4 * 4);
  float* tkS    = (float*)carve((size_t)B * PRE_NMS * 4);
  float* tkB    = (float*)carve((size_t)B * PRE_NMS * 4 * 4);
  float* out    = (float*)d_out;

  reorg_weights<<<(9 * 256 * 256 + 255) / 256, 256, 0, stream>>>(
      w_stem, wB, 9 * 256 * 256);

  for (int l = 0; l < 3; ++l) {
    const int H = Hs[l], W = Ws[l], HW = H * W;
    const int total3 = B * HW * 3;
    init_head<<<(total3 + 255) / 256, 256, 0, stream>>>(scores, deltas, b_obj,
                                                        b_box, total3);
    dim3 grid((B * HW) / 64, 2);
    stem_gemm_fused<<<grid, 256, 0, stream>>>(feats[l], wB, b_stem, w_obj,
                                              w_box, scores, deltas, H, W);
    decode_boxes<<<(total3 + 255) / 256, 256, 0, stream>>>(
        deltas, boxes, strides[l], strides[0], HW, W, total3, Hs[0]);
    topk400<<<B, 1024, 0, stream>>>(scores, boxes, HW * 3, tkS, tkB);
    nms100<<<B, 512, 0, stream>>>(tkB, out, l);
  }
}
